// Generator_11665131176433
// MI455X (gfx1250) — compile-verified, matrix operand order checked by reading, not run on hardware
//
#include <hip/hip_runtime.h>

// ---------------- problem constants ----------------
#define BB 32
#define NN 2048
#define KNBR 16
#define MM 2025          // 45*45
#define GRIDV 10.0f      // linspace(10,10,45) is constant 10.0

typedef __attribute__((ext_vector_type(16))) _Float16 v16h;
typedef __attribute__((ext_vector_type(8)))  float    v8f;

// ---------------- helpers ----------------
__device__ inline void atomicMaxFloat(float* addr, float val) {
    // requires init to -inf (0xFF800000)
    if (val >= 0.0f) atomicMax((int*)addr, __float_as_int(val));
    else             atomicMin((unsigned int*)addr, __float_as_uint(val));
}

// CDNA5 async global->LDS copy, 16 bytes per lane, tracked by ASYNCcnt.
__device__ inline void async_b128(void* lds_ptr, const void* gptr) {
    unsigned lds = (unsigned)(unsigned long long)lds_ptr;   // LDS byte address (addr[31:0])
    asm volatile("global_load_async_to_lds_b128 %0, %1, off"
                 :: "v"(lds), "v"(gptr)
                 : "memory");
}
// Wait until this wave's outstanding async count <= n (n is block-uniform).
__device__ inline void async_wait_n(int n) {
    if (n == 0)      asm volatile("s_wait_asynccnt 0x0" ::: "memory");
    else if (n == 1) asm volatile("s_wait_asynccnt 0x1" ::: "memory");
    else if (n == 2) asm volatile("s_wait_asynccnt 0x2" ::: "memory");
    else             asm volatile("s_wait_asynccnt 0x3" ::: "memory");
}

// Load a 16x32 f16 fragment row (A or B operand) from a row-major LDS tile.
// Per ISA layout: lanes 0-15 hold K=[kb..kb+7] in v0..3 and K=[kb+16..kb+23] in v4..7.
__device__ inline v16h load_frag(const _Float16* rowbase, int kb) {
    v16h f = {};
#pragma unroll
    for (int i = 0; i < 8; ++i) {
        f[i]     = rowbase[kb + i];
        f[i + 8] = rowbase[kb + 16 + i];
    }
    return f;
}

// ---------------- weight convert: fp32 [dout,din] -> f16 [dout,dinp], zero padded ----------------
__global__ void cvt_w_kernel(const float* __restrict__ src, _Float16* __restrict__ dst,
                             int dout, int din, int dinp) {
    int i = blockIdx.x * blockDim.x + threadIdx.x;
    if (i >= dout * dinp) return;
    int o = i / dinp, k = i % dinp;
    dst[i] = (_Float16)((k < din) ? src[(size_t)o * din + k] : 0.0f);
}

// ---------------- squared-norm rows (fp32) ----------------
__global__ void rowsum2_kernel(const float* __restrict__ p, float* __restrict__ o, int total) {
    int i = blockIdx.x * blockDim.x + threadIdx.x;
    if (i < total) {
        float x = p[3 * (size_t)i + 0], y = p[3 * (size_t)i + 1], z = p[3 * (size_t)i + 2];
        o[i] = x * x + y * y + z * z;
    }
}

// ---------------- KNN: top-16 smallest distances (self included, sorted ascending) ----------------
__global__ __launch_bounds__(128) void knn_kernel(const float* __restrict__ pts,
                                                  const float* __restrict__ xx,
                                                  int* __restrict__ idx) {
    int b = blockIdx.y;
    int n = blockIdx.x * 128 + threadIdx.x;   // N % 128 == 0
    const float* pb = pts + (size_t)b * NN * 3;
    float px = pb[3 * n + 0], py = pb[3 * n + 1], pz = pb[3 * n + 2];
    float xn = xx[(size_t)b * NN + n];

    float bd[KNBR]; int bi[KNBR];
#pragma unroll
    for (int k = 0; k < KNBR; ++k) { bd[k] = __builtin_inff(); bi[k] = 0; }

    __shared__ float sp[128][3];
    __shared__ float sx[128];
    for (int m0 = 0; m0 < NN; m0 += 128) {
        int mm = m0 + threadIdx.x;
        sp[threadIdx.x][0] = pb[3 * mm + 0];
        sp[threadIdx.x][1] = pb[3 * mm + 1];
        sp[threadIdx.x][2] = pb[3 * mm + 2];
        sx[threadIdx.x] = xx[(size_t)b * NN + mm];
        __syncthreads();
        for (int j = 0; j < 128; ++j) {
            float d = xn + sx[j] - 2.0f * (px * sp[j][0] + py * sp[j][1] + pz * sp[j][2]);
            if (d < bd[KNBR - 1]) {
                int t = KNBR - 1;
                while (t > 0 && bd[t - 1] > d) { bd[t] = bd[t - 1]; bi[t] = bi[t - 1]; --t; }
                bd[t] = d; bi[t] = m0 + j;
            }
        }
        __syncthreads();
    }
    int* op = idx + ((size_t)b * NN + n) * KNBR;
#pragma unroll
    for (int k = 0; k < KNBR; ++k) op[k] = bi[k];
}

// ---------------- x12 = [pts(3), outer(nb0, nb1)(9)] -> f16 ----------------
__global__ void build_x12_kernel(const float* __restrict__ pts, const int* __restrict__ idx,
                                 _Float16* __restrict__ x12) {
    int t = blockIdx.x * blockDim.x + threadIdx.x;   // over B*N
    if (t >= BB * NN) return;
    int b = t / NN;
    const float* pb = pts + (size_t)b * NN * 3;
    const int* id = idx + (size_t)t * KNBR;
    int i0 = id[0], i1 = id[1];
    float a0 = pb[3 * i0], a1 = pb[3 * i0 + 1], a2 = pb[3 * i0 + 2];
    float c0 = pb[3 * i1], c1 = pb[3 * i1 + 1], c2 = pb[3 * i1 + 2];
    _Float16* o = x12 + (size_t)t * 12;
    o[0] = (_Float16)pts[(size_t)t * 3 + 0];
    o[1] = (_Float16)pts[(size_t)t * 3 + 1];
    o[2] = (_Float16)pts[(size_t)t * 3 + 2];
    o[3] = (_Float16)(a0 * c0); o[4]  = (_Float16)(a0 * c1); o[5]  = (_Float16)(a0 * c2);
    o[6] = (_Float16)(a1 * c0); o[7]  = (_Float16)(a1 * c1); o[8]  = (_Float16)(a1 * c2);
    o[9] = (_Float16)(a2 * c0); o[10] = (_Float16)(a2 * c1); o[11] = (_Float16)(a2 * c2);
}

// ---------------- generic WMMA GEMM:  Y = act(X * W^T + b) ----------------
// IN_MODE: 0 = dense f16 X; 1 = [feat16(512)|10,10]; 2 = [feat16(512)|f1_16(3)]; 3 = dense f32 X
// OUT_MODE: 0 = f16 store; 1 = f32 store; 2 = fused global max into f32 Y[b*dout+n]
template <int IN_MODE>
__device__ inline _Float16 fetch_scalar(const _Float16* __restrict__ X16,
                                        const float* __restrict__ Xf32,
                                        const _Float16* __restrict__ feat,
                                        const _Float16* __restrict__ extra,
                                        int row, int k, int rows, int din, int Mrows) {
    if (row >= rows) return (_Float16)0.0f;
    if (IN_MODE == 0) return (k < din) ? X16[(size_t)row * din + k] : (_Float16)0.0f;
    if (IN_MODE == 3) return (_Float16)((k < din) ? Xf32[(size_t)row * din + k] : 0.0f);
    int b = row / Mrows;
    if (k < 512) return feat[(size_t)b * 512 + k];
    if (IN_MODE == 1) return (k < 514) ? (_Float16)GRIDV : (_Float16)0.0f;
    return (k < 515) ? extra[(size_t)row * 3 + (k - 512)] : (_Float16)0.0f;
}

#define TM 128
#define TN 64
#define TK 32

template <int IN_MODE, int OUT_MODE>
__global__ __launch_bounds__(256) void wmma_gemm_kernel(
    const _Float16* __restrict__ X16, const float* __restrict__ Xf32,
    const _Float16* __restrict__ W16, const float* __restrict__ bias,
    void* __restrict__ Yout, int rows, int din, int dinp, int dout, int relu,
    const _Float16* __restrict__ feat, const _Float16* __restrict__ extra,
    int Mrows, int Npts) {
    __shared__ __align__(16) _Float16 As[2][TM][40];   // double-buffered
    __shared__ __align__(16) _Float16 Bs[2][TN][40];

    const int tid = threadIdx.x;
    const int tileM = blockIdx.x * TM;
    const int tileN = blockIdx.y * TN;
    const int wave = tid >> 5;
    const int lane = tid & 31;

    const bool fullM = (tileM + TM) <= rows;
    const bool fullN = (tileN + TN) <= dout;

    v8f zero = {};
    v8f acc[4];
#pragma unroll
    for (int t = 0; t < 4; ++t) acc[t] = zero;

    const int ksteps = dinp >> 5;          // dinp is a multiple of 32

    // Stage one 128x32 A tile + 64x32 B tile into buffer bi.
    // Returns #async instructions issued by this wave (block-uniform).
    auto stage = [&](int ks, int bi) -> int {
        const int k0 = ks * TK;
        int cnt = 0;
        bool a_fast = fullM &&
                      ((IN_MODE == 0) ? (k0 + TK) <= din
                                      : (IN_MODE == 1 || IN_MODE == 2) && ((k0 + TK) <= 512));
        if (a_fast) {
#pragma unroll
            for (int s = 0; s < 2; ++s) {
                int slot = tid + s * 256;                 // 512 slots of 8 halves
                int r = slot >> 2, c = (slot & 3) * 8;
                const _Float16* g;
                if (IN_MODE == 0) {
                    g = X16 + (size_t)(tileM + r) * din + k0 + c;
                } else {
                    int b = (tileM + r) / Mrows;
                    g = feat + (size_t)b * 512 + k0 + c;
                }
                async_b128(&As[bi][r][c], g);
            }
            cnt += 2;
        } else {
            for (int i = tid; i < TM * TK; i += 256) {
                int r = i >> 5, c = i & 31;
                As[bi][r][c] = fetch_scalar<IN_MODE>(X16, Xf32, feat, extra,
                                                     tileM + r, k0 + c, rows, din, Mrows);
            }
        }
        if (fullN) {   // W16 rows are padded to dinp, so K is always full
            int r = tid >> 2, c = (tid & 3) * 8;          // 256 slots of 8 halves
            async_b128(&Bs[bi][r][c], W16 + (size_t)(tileN + r) * dinp + k0 + c);
            cnt += 1;
        } else {
            for (int i = tid; i < TN * TK; i += 256) {
                int r = i >> 5, c = i & 31;
                int o = tileN + r;
                Bs[bi][r][c] = (o < dout) ? W16[(size_t)o * dinp + k0 + c] : (_Float16)0.0f;
            }
        }
        return cnt;
    };

    const int arow = wave * 16 + (lane & 15);
    const int kb = (lane >> 4) * 8;

    stage(0, 0);
    for (int ks = 0; ks < ksteps; ++ks) {
        // prefetch next tile into the other buffer while current is in flight
        int nxt = 0;
        if (ks + 1 < ksteps) nxt = stage(ks + 1, (ks + 1) & 1);
        // async loads complete in order: waiting to <= nxt guarantees tile ks resident
        async_wait_n(nxt);
        __syncthreads();

        const _Float16(*A)[40] = As[ks & 1];
        const _Float16(*Bt)[40] = Bs[ks & 1];
        v16h afrag = load_frag(&A[arow][0], kb);
        v16h bf[4];
#pragma unroll
        for (int t = 0; t < 4; ++t) bf[t] = load_frag(&Bt[t * 16 + (lane & 15)][0], kb);
#pragma unroll
        for (int t = 0; t < 4; ++t)
            acc[t] = __builtin_amdgcn_wmma_f32_16x16x32_f16(
                false, afrag, false, bf[t], (short)0, acc[t], false, false);
        __syncthreads();
    }

    // ---- epilogue ----
    const int halfsel = lane >> 4;
    if (OUT_MODE == 2) {
        // fused global maxpool: reduce the 8 rows this thread holds, then one atomic
#pragma unroll
        for (int t = 0; t < 4; ++t) {
            int n = tileN + t * 16 + (lane & 15);
            float bv = (n < dout) ? bias[n] : 0.0f;
            float mx = -__builtin_inff();
#pragma unroll
            for (int r = 0; r < 8; ++r) {
                int m = tileM + wave * 16 + halfsel * 8 + r;
                float v = acc[t][r] + bv;
                if (relu) v = v > 0.0f ? v : 0.0f;
                if (m < rows) mx = fmaxf(mx, v);
            }
            int m0 = tileM + wave * 16 + halfsel * 8;
            if (m0 < rows && n < dout) {
                int b = m0 / Npts;   // all 8 rows share one batch (Npts % TM == 0)
                atomicMaxFloat(&((float*)Yout)[(size_t)b * dout + n], mx);
            }
        }
    } else {
#pragma unroll
        for (int t = 0; t < 4; ++t) {
            int n = tileN + t * 16 + (lane & 15);
            float bv = (n < dout) ? bias[n] : 0.0f;
#pragma unroll
            for (int r = 0; r < 8; ++r) {
                int m = tileM + wave * 16 + halfsel * 8 + r;
                float v = acc[t][r] + bv;
                if (relu) v = v > 0.0f ? v : 0.0f;
                if (m < rows && n < dout) {
                    if (OUT_MODE == 0) ((_Float16*)Yout)[(size_t)m * dout + n] = (_Float16)v;
                    else               ((float*)Yout)[(size_t)m * dout + n] = v;
                }
            }
        }
    }
}

// ---------------- gather + max over K neighbors (f16) ----------------
__global__ void gathermax_kernel(const _Float16* __restrict__ fin, const int* __restrict__ idx,
                                 _Float16* __restrict__ fout, int C) {
    size_t t = (size_t)blockIdx.x * blockDim.x + threadIdx.x;
    size_t total = (size_t)BB * NN * C;
    if (t >= total) return;
    int c = (int)(t % C);
    size_t bn = t / C;
    int b = (int)(bn / NN);
    const int* id = idx + bn * KNBR;
    float mx = -__builtin_inff();
#pragma unroll
    for (int k = 0; k < KNBR; ++k) {
        float v = (float)fin[((size_t)b * NN + id[k]) * C + c];
        mx = fmaxf(mx, v);
    }
    fout[bn * C + c] = (_Float16)mx;
}

__global__ void fill_neginf_kernel(float* p, int n) {
    int i = blockIdx.x * blockDim.x + threadIdx.x;
    if (i < n) p[i] = -__builtin_inff();
}

// ---------------- chamfer: per (b,i in Na) min over j in Nb (fp32) ----------------
__global__ __launch_bounds__(128) void chamfer_min_kernel(
    const float* __restrict__ A, const float* __restrict__ aa,
    const float* __restrict__ Bp, const float* __restrict__ bb,
    float* __restrict__ outmin, int Na, int Nb) {
    int b = blockIdx.y;
    int i = blockIdx.x * 128 + threadIdx.x;
    bool valid = i < Na;
    float ax = 0, ay = 0, az = 0, aq = 0;
    if (valid) {
        ax = A[((size_t)b * Na + i) * 3 + 0];
        ay = A[((size_t)b * Na + i) * 3 + 1];
        az = A[((size_t)b * Na + i) * 3 + 2];
        aq = aa[(size_t)b * Na + i];
    }
    __shared__ float sp[128][3];
    __shared__ float sq[128];
    float best = __builtin_inff();
    for (int j0 = 0; j0 < Nb; j0 += 128) {
        int j = j0 + threadIdx.x;
        if (j < Nb) {
            sp[threadIdx.x][0] = Bp[((size_t)b * Nb + j) * 3 + 0];
            sp[threadIdx.x][1] = Bp[((size_t)b * Nb + j) * 3 + 1];
            sp[threadIdx.x][2] = Bp[((size_t)b * Nb + j) * 3 + 2];
            sq[threadIdx.x] = bb[(size_t)b * Nb + j];
        } else {
            sp[threadIdx.x][0] = 0; sp[threadIdx.x][1] = 0; sp[threadIdx.x][2] = 0;
            sq[threadIdx.x] = __builtin_inff();
        }
        __syncthreads();
        for (int jj = 0; jj < 128; ++jj) {
            float d = aq + sq[jj] - 2.0f * (ax * sp[jj][0] + ay * sp[jj][1] + az * sp[jj][2]);
            best = fminf(best, d);
        }
        __syncthreads();
    }
    if (valid) outmin[(size_t)b * Na + i] = best;
}

// ---------------- final loss reduce (single block) ----------------
__global__ __launch_bounds__(256) void loss_kernel(const float* __restrict__ rowmin,
                                                   const float* __restrict__ colmin,
                                                   float* __restrict__ out) {
    __shared__ float sr[256], sc[256];
    float s1 = 0, s2 = 0;
    for (int i = threadIdx.x; i < BB * NN; i += 256) s1 += rowmin[i];
    for (int i = threadIdx.x; i < BB * MM; i += 256) s2 += colmin[i];
    sr[threadIdx.x] = s1; sc[threadIdx.x] = s2;
    __syncthreads();
    for (int s = 128; s > 0; s >>= 1) {
        if (threadIdx.x < s) { sr[threadIdx.x] += sr[threadIdx.x + s]; sc[threadIdx.x] += sc[threadIdx.x + s]; }
        __syncthreads();
    }
    if (threadIdx.x == 0)
        out[0] = sc[0] / (float)(BB * MM) + sr[0] / (float)(BB * NN);
}

// ---------------- write outputs: transpose [B,M,3]->[B,3,M], copy feat ----------------
__global__ void writeout_kernel(const float* __restrict__ outp, const _Float16* __restrict__ feat,
                                float* __restrict__ dout) {
    int t = blockIdx.x * blockDim.x + threadIdx.x;
    const int n1 = BB * 3 * MM;
    if (t < n1) {
        int b = t / (3 * MM);
        int rem = t % (3 * MM);
        int c = rem / MM;
        int m = rem % MM;
        dout[t] = outp[((size_t)b * MM + m) * 3 + c];
    } else if (t < n1 + BB * 512) {
        dout[t] = (float)feat[t - n1];
    }
}

// ---------------- host launch ----------------
struct Layer { const float* w; const float* b; };

extern "C" void kernel_launch(void* const* d_in, const int* in_sizes, int n_in,
                              void* d_out, int out_size, void* d_ws, size_t ws_size,
                              hipStream_t stream) {
    (void)in_sizes; (void)n_in; (void)out_size; (void)ws_size;
    const float* pts = (const float*)d_in[0];
    Layer L[15];
    for (int i = 0; i < 15; ++i) { L[i].w = (const float*)d_in[1 + 2 * i]; L[i].b = (const float*)d_in[2 + 2 * i]; }
    enum { m1a = 0, m1b, m1c, lin1, conv1, lin2, conv2, m2a, m2b, f1a, f1b, f1c, f2a, f2b, f2c };
    const int DIN[15]  = {12, 64, 64, 64, 64, 128, 128, 1024, 512, 514, 512, 512, 515, 512, 512};
    const int DOUT[15] = {64, 64, 64, 64, 128, 128, 1024, 512, 512, 512, 512, 3, 512, 512, 3};

    char* ws = (char*)d_ws;
    size_t cur = 0;
    auto alloc = [&](size_t bytes) -> void* {
        cur = (cur + 255) & ~(size_t)255;
        void* p = ws + cur;
        cur += bytes;
        return p;
    };

    // padded f16 weights
    _Float16* w16[15];
    int DINP[15];
    for (int i = 0; i < 15; ++i) {
        DINP[i] = (DIN[i] + 31) & ~31;
        w16[i] = (_Float16*)alloc((size_t)DOUT[i] * DINP[i] * sizeof(_Float16));
    }

    float*    xx     = (float*)alloc((size_t)BB * NN * sizeof(float));
    float*    yy     = (float*)alloc((size_t)BB * MM * sizeof(float));
    int*      idx    = (int*)  alloc((size_t)BB * NN * KNBR * sizeof(int));
    _Float16* x12    = (_Float16*)alloc((size_t)BB * NN * 12 * sizeof(_Float16));
    _Float16* encA   = (_Float16*)alloc((size_t)BB * NN * 128 * sizeof(_Float16));
    _Float16* encB   = (_Float16*)alloc((size_t)BB * NN * 128 * sizeof(_Float16));
    float*    gmax   = (float*)alloc((size_t)BB * 1024 * sizeof(float));
    _Float16* m2tmp  = (_Float16*)alloc((size_t)BB * 512 * sizeof(_Float16));
    _Float16* feat   = (_Float16*)alloc((size_t)BB * 512 * sizeof(_Float16));
    _Float16* decA   = (_Float16*)alloc((size_t)BB * MM * 512 * sizeof(_Float16));
    _Float16* decB   = (_Float16*)alloc((size_t)BB * MM * 512 * sizeof(_Float16));
    _Float16* f1buf  = (_Float16*)alloc((size_t)BB * MM * 3 * sizeof(_Float16));
    float*    outp   = (float*)alloc((size_t)BB * MM * 3 * sizeof(float));
    float*    rowmin = (float*)alloc((size_t)BB * NN * sizeof(float));
    float*    colmin = (float*)alloc((size_t)BB * MM * sizeof(float));

    // convert weights once
    for (int i = 0; i < 15; ++i) {
        int n = DOUT[i] * DINP[i];
        cvt_w_kernel<<<(n + 255) / 256, 256, 0, stream>>>(L[i].w, w16[i], DOUT[i], DIN[i], DINP[i]);
    }

    auto gemm = [&](int in_mode, int out_mode, const _Float16* X16, const float* Xf32,
                    int li, void* Y, int rows, int relu,
                    const _Float16* featp, const _Float16* extra) {
        int din = DIN[li], dinp = DINP[li], dout = DOUT[li];
        dim3 grid((rows + TM - 1) / TM, (dout + TN - 1) / TN);
        const _Float16* W = w16[li];
        const float* bias = L[li].b;
        if (in_mode == 0 && out_mode == 0)
            wmma_gemm_kernel<0, 0><<<grid, 256, 0, stream>>>(X16, Xf32, W, bias, Y, rows, din, dinp, dout, relu, featp, extra, MM, NN);
        else if (in_mode == 0 && out_mode == 1)
            wmma_gemm_kernel<0, 1><<<grid, 256, 0, stream>>>(X16, Xf32, W, bias, Y, rows, din, dinp, dout, relu, featp, extra, MM, NN);
        else if (in_mode == 0 && out_mode == 2)
            wmma_gemm_kernel<0, 2><<<grid, 256, 0, stream>>>(X16, Xf32, W, bias, Y, rows, din, dinp, dout, relu, featp, extra, MM, NN);
        else if (in_mode == 1)
            wmma_gemm_kernel<1, 0><<<grid, 256, 0, stream>>>(X16, Xf32, W, bias, Y, rows, din, dinp, dout, relu, featp, extra, MM, NN);
        else if (in_mode == 2)
            wmma_gemm_kernel<2, 0><<<grid, 256, 0, stream>>>(X16, Xf32, W, bias, Y, rows, din, dinp, dout, relu, featp, extra, MM, NN);
        else
            wmma_gemm_kernel<3, 0><<<grid, 256, 0, stream>>>(X16, Xf32, W, bias, Y, rows, din, dinp, dout, relu, featp, extra, MM, NN);
    };

    const int RN = BB * NN;   // 65536
    const int RM = BB * MM;   // 64800

    // ---- encoder ----
    rowsum2_kernel<<<(RN + 255) / 256, 256, 0, stream>>>(pts, xx, RN);
    { dim3 g(NN / 128, BB); knn_kernel<<<g, 128, 0, stream>>>(pts, xx, idx); }
    build_x12_kernel<<<(RN + 255) / 256, 256, 0, stream>>>(pts, idx, x12);

    gemm(0, 0, x12,  nullptr, m1a, encA, RN, 1, nullptr, nullptr);
    gemm(0, 0, encA, nullptr, m1b, encB, RN, 1, nullptr, nullptr);
    gemm(0, 0, encB, nullptr, m1c, encA, RN, 1, nullptr, nullptr);
    gathermax_kernel<<<(int)(((size_t)RN * 64 + 255) / 256), 256, 0, stream>>>(encA, idx, encB, 64);
    gemm(0, 0, encB, nullptr, lin1,  encA, RN, 0, nullptr, nullptr);
    gemm(0, 0, encA, nullptr, conv1, encB, RN, 1, nullptr, nullptr);
    gathermax_kernel<<<(int)(((size_t)RN * 128 + 255) / 256), 256, 0, stream>>>(encB, idx, encA, 128);
    gemm(0, 0, encA, nullptr, lin2, encB, RN, 0, nullptr, nullptr);
    // conv2 fused with global maxpool over N
    fill_neginf_kernel<<<(BB * 1024 + 255) / 256, 256, 0, stream>>>(gmax, BB * 1024);
    gemm(0, 2, encB, nullptr, conv2, gmax, RN, 0, nullptr, nullptr);
    gemm(3, 0, nullptr, gmax, m2a, m2tmp, BB, 1, nullptr, nullptr);
    gemm(0, 0, m2tmp, nullptr, m2b, feat, BB, 0, nullptr, nullptr);

    // ---- folding decoder (grid = constant 10.0, inputs synthesized in GEMM loader) ----
    gemm(1, 0, nullptr, nullptr, f1a, decA, RM, 1, feat, nullptr);
    gemm(0, 0, decA,   nullptr, f1b, decB, RM, 1, nullptr, nullptr);
    gemm(0, 0, decB,   nullptr, f1c, f1buf, RM, 0, nullptr, nullptr);
    gemm(2, 0, nullptr, nullptr, f2a, decA, RM, 1, feat, f1buf);
    gemm(0, 0, decA,   nullptr, f2b, decB, RM, 1, nullptr, nullptr);
    gemm(0, 1, decB,   nullptr, f2c, outp, RM, 0, nullptr, nullptr);

    // ---- chamfer ----
    rowsum2_kernel<<<(RM + 255) / 256, 256, 0, stream>>>(outp, yy, RM);
    { dim3 g((NN + 127) / 128, BB); chamfer_min_kernel<<<g, 128, 0, stream>>>(pts, xx, outp, yy, rowmin, NN, MM); }
    { dim3 g((MM + 127) / 128, BB); chamfer_min_kernel<<<g, 128, 0, stream>>>(outp, yy, pts, xx, colmin, MM, NN); }

    // ---- outputs ----
    float* dout = (float*)d_out;
    const int n_out01 = BB * 3 * MM + BB * 512;
    writeout_kernel<<<(n_out01 + 255) / 256, 256, 0, stream>>>(outp, feat, dout);
    loss_kernel<<<1, 256, 0, stream>>>(rowmin, colmin, dout + n_out01);
}
